// First_n_Last_9225589751956
// MI455X (gfx1250) — matrix operand order repaired and checked by reading, not compiled, not run on hardware
//
#include <hip/hip_runtime.h>

typedef _Float16 v8h  __attribute__((ext_vector_type(8)));
typedef _Float16 v16h __attribute__((ext_vector_type(16)));
typedef float    v8f  __attribute__((ext_vector_type(8)));
typedef unsigned int u32x4 __attribute__((ext_vector_type(4)));
typedef int          i32x4 __attribute__((ext_vector_type(4)));
typedef int          i32x8 __attribute__((ext_vector_type(8)));

// Problem constants (fixed by the reference setup_inputs()).
constexpr int BATCH = 4;
constexpr int SEQ   = 2048;
constexpr int HID   = 768;            // H (output cols)
constexpr int NSPAN = SEQ * 8;        // spans per batch = 16384 (pow2)
constexpr int KDIM  = 2 * HID;        // 1536
constexpr int MTOT  = BATCH * NSPAN;  // 65536 GEMM rows

constexpr int BM = 64;                // spans per block
constexpr int KC = 32;                // K-chunk per TDM transfer
constexpr int NCHUNK = KDIM / KC;     // 48
constexpr int CHUNK_HALVES = HID * KC;        // 24576 halves = 48 KB
constexpr int TILE0 = CHUNK_HALVES / 4;       // 6144 8-byte TDM elements

// LDS strides (halves). ASTR rows: 16B aligned + bank-skewed (+4 banks/row).
// BSTR = 40: TDM pads 4 DWORDs after every 16 DWORDs -> 80B rows, 16B aligned,
// 20 dwords/row -> the 16 fragment row-lanes hit 16 distinct bank groups.
constexpr int ASTR = KDIM + 8;        // 1544
constexpr int BSTR = KC + 8;          // 40

static __device__ inline v16h ld_frag(const _Float16* p0, const _Float16* p1) {
    v8h lo = *(const v8h*)p0;
    v8h hi = *(const v8h*)p1;
    return __builtin_shufflevector(lo, hi, 0,1,2,3,4,5,6,7,8,9,10,11,12,13,14,15);
}

// ---------------------------------------------------------------------------
// Prep: W (1536x768 f32) -> Wh f16, chunk-linear: Wh[(c*768+n)*32+kk] with
// k = c*32+kk. Destination-linear so the 2B stores are fully coalesced.
// ---------------------------------------------------------------------------
__global__ __launch_bounds__(256)
void w_to_f16_frag(const float* __restrict__ W, _Float16* __restrict__ Wh) {
    const int gid   = blockIdx.x * 256 + threadIdx.x;   // < 48*768*32
    const int chunk = gid / (HID * KC);
    const int r     = gid - chunk * (HID * KC);
    const int nn    = r >> 5;
    const int kk    = r & 31;
    Wh[gid] = (_Float16)W[(size_t)(chunk * KC + kk) * HID + nn];
}

// Issue one TDM transfer: 48KB dense global chunk -> padded LDS image.
// D# per CDNA5 ISA sect. 8 (1-D tile, 8B elements).
static __device__ inline void tdm_load_chunk(const _Float16* gsrc, unsigned lds_off) {
#if __has_builtin(__builtin_amdgcn_tensor_load_to_lds)
    const unsigned long long ga = (unsigned long long)(size_t)gsrc;
    u32x4 g0;
    g0[0] = 1u;                                        // count=1 (valid), user mode
    g0[1] = lds_off;                                   // lds_addr
    g0[2] = (unsigned)(ga & 0xFFFFFFFFu);              // global_addr[31:0]
    g0[3] = (unsigned)((ga >> 32) & 0x01FFFFFFu)       // global_addr[56:32]
          | (2u << 30);                                // type = 2 (image)
    i32x8 g1;
    g1[0] = (int)((3u << 16)      // data_size = 8B
                | (1u << 20)      // pad_enable
                | (3u << 22)      // pad_interval: 16 DWORDs (one 64B row)
                | (3u << 25));    // pad_amount: 4 DWORDs (16B skew pad)
    g1[1] = (int)(((unsigned)TILE0 & 0xFFFFu) << 16);  // tensor_dim0 lo16
    g1[2] = (int)(((unsigned)TILE0 >> 16) | (1u << 16)); // tensor_dim0 hi16, tensor_dim1=1
    g1[3] = (int)(((unsigned)TILE0 & 0xFFFFu) << 16);  // tile_dim0 = 6144
    g1[4] = 0;                                         // tile_dim1=0, tile_dim2=0
    g1[5] = (int)TILE0;                                // tensor_dim0_stride lo32
    g1[6] = 0;
    g1[7] = 0;
    i32x4 gz = {0, 0, 0, 0};
#if __clang_major__ >= 23
    i32x8 gz8 = {};
    __builtin_amdgcn_tensor_load_to_lds(g0, g1, gz, gz, gz8, 0);
#else
    __builtin_amdgcn_tensor_load_to_lds(g0, g1, gz, gz, 0);
#endif
#endif
}

// ---------------------------------------------------------------------------
// Fused: span start/end window max-pool (f16, LDS) + WMMA GEMM vs Wh + bias.
// Block: 64 spans x 768 cols, 8 waves; TDM double-buffers the W chunks.
// ---------------------------------------------------------------------------
__global__ __launch_bounds__(256, 1)
void span_firstlast_wmma(const float* __restrict__ tok,      // (B,S,H) f32
                         const int*   __restrict__ span_ids, // (B,N,2)
                         const int*   __restrict__ span_mask,// (B,N)
                         const int*   __restrict__ pooling,  // scalar
                         const _Float16* __restrict__ Wh,    // prepped f16 W
                         const float* __restrict__ bias,     // (H,)
                         float* __restrict__ out)            // (B,N,H) f32
{
    __shared__ __align__(16) _Float16 As[BM * ASTR];        // 193 KB
    __shared__ __align__(16) _Float16 Bs[2][HID * BSTR];    // 2 x 60 KB

    const int t    = threadIdx.x;
    const int lane = t & 31;
    const int wave = t >> 5;        // 0..7
    const int wr   = wave >> 2;     // 0..1 : 32-row half of the M tile
    const int wc   = wave & 3;      // 0..3 : 192-col strip
    const int lc   = lane & 15;
    const int gh   = lane >> 4;     // lane-half -> K-group per ISA layout

    const int m0 = blockIdx.x * BM;
    const unsigned bsoff[2] = { (unsigned)(size_t)(void*)&Bs[0][0],
                                (unsigned)(size_t)(void*)&Bs[1][0] };

#if __has_builtin(__builtin_amdgcn_tensor_load_to_lds)
    // Kick chunk 0 on the TDM before pooling so it overlaps the gather phase.
    if (wave == 0) tdm_load_chunk(Wh, bsoff[0]);
#endif

    const int window = (pooling[0] != 0) ? 1 : 3;

    // ---------------- Phase A: gather + masked window max -> As (f16) -------
    for (int s = 0; s < BM; ++s) {
        const int m   = m0 + s;
        const int b   = m >> 14;                 // m / NSPAN
        const int st  = span_ids[2 * m];
        const int en  = span_ids[2 * m + 1];
        const bool ok = span_mask[m] != 0;
        int cnt = en - st;
        if (cnt > window) cnt = window;
        const long tokrow = (long)b * SEQ;

        for (int d = t; d < KDIM; d += 256) {
            float v = 0.0f;
            if (ok) {
                int hh, base;
                if (d < HID) { hh = d;       base = st; }        // start window
                else         { hh = d - HID; base = en - cnt; }  // end window
                const float* p = tok + (tokrow + base) * (long)HID + hh;
                v = p[0];
                for (int j = 1; j < cnt; ++j) v = fmaxf(v, p[(long)j * HID]);
            }
            As[s * ASTR + d] = (_Float16)v;
        }
    }

#if !__has_builtin(__builtin_amdgcn_tensor_load_to_lds)
    // Fallback: synchronous staged copy of chunk 0 (same padded image).
    for (int g = t; g < HID * (KC / 8); g += 256) {
        const int nn = g >> 2, k8 = (g & 3) * 8;
        *(v8h*)&Bs[0][nn * BSTR + k8] = *(const v8h*)(Wh + nn * KC + k8);
    }
#endif

    v8f acc0[12];
    v8f acc1[12];
    #pragma unroll
    for (int tt = 0; tt < 12; ++tt) { acc0[tt] = (v8f){}; acc1[tt] = (v8f){}; }

    const _Float16* arow0 = &As[(wr * 32 + lc) * ASTR];
    const _Float16* arow1 = &As[(wr * 32 + 16 + lc) * ASTR];

    for (int c = 0; c < NCHUNK; ++c) {
        const int cur = c & 1;
#if __has_builtin(__builtin_amdgcn_tensor_load_to_lds)
        if (wave == 0) {
            if (c + 1 < NCHUNK) {   // prefetch next chunk on the TDM
                tdm_load_chunk(Wh + (size_t)(c + 1) * CHUNK_HALVES, bsoff[cur ^ 1]);
                __builtin_amdgcn_s_wait_tensorcnt(1);  // chunk c landed
            } else {
                __builtin_amdgcn_s_wait_tensorcnt(0);
            }
        }
#else
        if (c + 1 < NCHUNK) {
            const _Float16* wsrc = Wh + (size_t)(c + 1) * CHUNK_HALVES;
            for (int g = t; g < HID * (KC / 8); g += 256) {
                const int nn = g >> 2, k8 = (g & 3) * 8;
                *(v8h*)&Bs[cur ^ 1][nn * BSTR + k8] = *(const v8h*)(wsrc + nn * KC + k8);
            }
        }
#endif
        __syncthreads();   // chunk c visible to all waves (and As on c==0)

        const int kb = c * KC;
        // A 16x32 f16 fragments (ISA 7.12.2): two 16B K-group loads each.
        v16h a0 = ld_frag(arow0 + kb + gh * 8, arow0 + kb + 16 + gh * 8);
        v16h a1 = ld_frag(arow1 + kb + gh * 8, arow1 + kb + 16 + gh * 8);
        const _Float16* bbase = &Bs[cur][(wc * 192 + lc) * BSTR + gh * 8];
        #pragma unroll
        for (int tt = 0; tt < 12; ++tt) {
            // B 32x16: lane = column, two 8-half K-groups per lane (same as A).
            const _Float16* bp = bbase + tt * (16 * BSTR);
            v16h b = ld_frag(bp, bp + 16);
            acc0[tt] = __builtin_amdgcn_wmma_f32_16x16x32_f16(
                           false, a0, false, b, (short)0, acc0[tt], false, false);
            acc1[tt] = __builtin_amdgcn_wmma_f32_16x16x32_f16(
                           false, a1, false, b, (short)0, acc1[tt], false, false);
        }
        __syncthreads();   // all reads of Bs[cur] done before it is refilled
    }

    // ---------------- Epilogue: bias add + store (ISA C/D layout) -----------
    #pragma unroll
    for (int tt = 0; tt < 12; ++tt) {
        const int col = wc * 192 + tt * 16 + lc;
        const float bv = bias[col];
        const int r0 = m0 + wr * 32 + gh * 8;        // row-tile 0
        const int r1 = r0 + 16;                      // row-tile 1
        #pragma unroll
        for (int j = 0; j < 8; ++j) {
            out[(size_t)(r0 + j) * HID + col] = acc0[tt][j] + bv;
            out[(size_t)(r1 + j) * HID + col] = acc1[tt][j] + bv;
        }
    }
}

extern "C" void kernel_launch(void* const* d_in, const int* in_sizes, int n_in,
                              void* d_out, int out_size, void* d_ws, size_t ws_size,
                              hipStream_t stream) {
    const float* tok  = (const float*)d_in[0];
    const int*   sid  = (const int*)  d_in[1];
    const int*   smk  = (const int*)  d_in[2];
    const int*   pool = (const int*)  d_in[3];
    const float* W    = (const float*)d_in[4];
    const float* bia  = (const float*)d_in[5];
    float*       outp = (float*)d_out;
    _Float16*    Wh   = (_Float16*)d_ws;   // 48*768*32*2 = 2.25 MB

    // 1) one-time W f32 -> f16 chunk-linear repack
    w_to_f16_frag<<<dim3((NCHUNK * HID * KC) / 256), dim3(256), 0, stream>>>(W, Wh);
    // 2) fused pooling + TDM-fed WMMA GEMM
    span_firstlast_wmma<<<dim3(MTOT / BM), dim3(256), 0, stream>>>(
        tok, sid, smk, pool, Wh, bia, outp);
}
